// ProposalModel_77678778515840
// MI455X (gfx1250) — matrix operand order repaired and checked by reading, not compiled, not run on hardware
//
#include <hip/hip_runtime.h>

typedef __attribute__((ext_vector_type(8)))  float   v8f;
typedef __attribute__((ext_vector_type(4)))  float   v4f;
typedef __attribute__((ext_vector_type(16))) __bf16  v16bf;
typedef __attribute__((ext_vector_type(8)))  __bf16  v8bf;
typedef __attribute__((ext_vector_type(4)))  unsigned int u32x4;
typedef __attribute__((ext_vector_type(8)))  int     i32x8;
typedef __attribute__((ext_vector_type(4)))  int     i32x4;
typedef unsigned short u16;
typedef __attribute__((ext_vector_type(4)))  u16     u16x4;

#define T_STEPS 2048
#define BATCH   32
#define FEAT_D  512
#define HID_D   512
#define GATE4   2048   // 4*HID
#define ANCH    64

// ---------- helpers ----------
__device__ __forceinline__ u16 f2bf_u(float f) {
  return __builtin_bit_cast(u16, (__bf16)f);   // native v_cvt, RNE
}
// branch-free sigmoid / tanh on the hardware transcendental pipe
__device__ __forceinline__ float sigmf(float x) {
  return __builtin_amdgcn_rcpf(1.0f + __expf(-x));
}
__device__ __forceinline__ float tanhfast(float x) {
  float e = __expf(-2.0f * x);
  return (1.0f - e) * __builtin_amdgcn_rcpf(1.0f + e);
}

// ---------- weight transpose + fp32 -> bf16 ----------
// in: (K,N) row-major fp32 ; out: (N,K) row-major bf16  (so B fragments are contiguous in K)
__global__ void k_transpose_to_bf16(const float* __restrict__ in, u16* __restrict__ out,
                                    int K, int N) {
  int total = K * N;
  for (int idx = blockIdx.x * blockDim.x + threadIdx.x; idx < total;
       idx += gridDim.x * blockDim.x) {
    int n = idx / K, k = idx - n * K;
    out[idx] = f2bf_u(in[k * N + n]);
  }
}

__global__ void k_init_counters(unsigned* p) {
  if (threadIdx.x < 8) p[threadIdx.x] = 0u;
}

// ---------- big parallel GEMM: C = A(M,K) fp32 * Wt(N,K) bf16 (+bias) ----------
// Compile-time M,N,K so strides fold into immediate offsets.
// STORE_T=true stores C transposed as (N,M): the WMMA C fragment is per-lane
// contiguous along M, so the epilogue becomes 2x global_store_b128 per sub-tile.
template <int M, int N, int K, bool STORE_T>
__global__ __launch_bounds__(256) void k_gemm_bf16(const float* __restrict__ A,
                                                   const u16* __restrict__ Wt,
                                                   const float* __restrict__ bias,
                                                   float* __restrict__ C) {
  constexpr int tilesN = N / 64;
  const int lane = threadIdx.x & 31;
  const int wave = threadIdx.x >> 5;
  const int gt = blockIdx.x * 8 + wave;
  const int tm = gt / tilesN;
  const int tn = gt - tm * tilesN;
  if (tm * 16 >= M) return;
  const int hlf = lane >> 4;        // which 16-lane half
  const int l16 = lane & 15;

  v8f acc[4];
#pragma unroll
  for (int j = 0; j < 4; ++j) {
    float bv = bias ? bias[tn * 64 + j * 16 + l16] : 0.0f;
#pragma unroll
    for (int r = 0; r < 8; ++r) acc[j][r] = bv;
  }

  const float* Arow  = A + (size_t)(tm * 16 + l16) * K;
  const u16*   Wbase = Wt + (size_t)(tn * 64 + l16) * K;

#pragma unroll 4
  for (int kc = 0; kc < K; kc += 32) {
    // A fragment (16x32 bf16): lane half h holds K = {h*8..h*8+7} U {16+h*8..16+h*8+7}
    v16bf af;
    {
      v4f a0 = *(const v4f*)(Arow + kc + hlf * 8);
      v4f a1 = *(const v4f*)(Arow + kc + hlf * 8 + 4);
      v4f a2 = *(const v4f*)(Arow + kc + 16 + hlf * 8);
      v4f a3 = *(const v4f*)(Arow + kc + 16 + hlf * 8 + 4);
#pragma unroll
      for (int i = 0; i < 4; ++i) {
        af[i]      = (__bf16)a0[i];
        af[4 + i]  = (__bf16)a1[i];
        af[8 + i]  = (__bf16)a2[i];
        af[12 + i] = (__bf16)a3[i];
      }
    }
#pragma unroll
    for (int j = 0; j < 4; ++j) {
      // B fragment (32x16 bf16): lane half h holds K = {h*8..h*8+7} U {16+h*8..16+h*8+7} of its column
      const u16* Bc = Wbase + j * 16 * K + kc;
      v8bf b0 = *(const v8bf*)(Bc + hlf * 8);
      v8bf b1 = *(const v8bf*)(Bc + 16 + hlf * 8);
      v16bf bf;
#pragma unroll
      for (int i = 0; i < 8; ++i) { bf[i] = b0[i]; bf[8 + i] = b1[i]; }
      acc[j] = __builtin_amdgcn_wmma_f32_16x16x32_bf16(false, af, false, bf,
                                                       (short)0, acc[j], false, false);
    }
  }

  // C/D layout: lane holds one column; lanes 0-15 rows 0-7, lanes 16-31 rows 8-15
  if (STORE_T) {
    float* Cp = C + (size_t)(tn * 64 + l16) * M + tm * 16 + hlf * 8;
#pragma unroll
    for (int j = 0; j < 4; ++j) {
      v4f lo, hi;
#pragma unroll
      for (int r = 0; r < 4; ++r) { lo[r] = acc[j][r]; hi[r] = acc[j][4 + r]; }
      float* p = Cp + (size_t)j * 16 * M;
      *(v4f*)p = lo;
      *(v4f*)(p + 4) = hi;
    }
  } else {
    float* Cp = C + (size_t)(tm * 16 + hlf * 8) * N + tn * 64 + l16;
#pragma unroll
    for (int j = 0; j < 4; ++j)
#pragma unroll
      for (int r = 0; r < 8; ++r)
        Cp[r * N + j * 16] = acc[j][r];
  }
}

// ---------- sequential LSTM scan (persistent, 16 WGs, device-wide barrier/step) ----------
// wiT: (4H, T*B) fp32 transposed input projection; Whh_t: (4H, H) bf16 (N,K layout);
// bias: (4H,); length: (B,) int64; h_out: (T,B,H) fp32; h_shared: (B,H) bf16 exchange buf.
__global__ __launch_bounds__(256, 1) void k_lstm_scan(const float* __restrict__ wiT,
                                                      const u16* __restrict__ Whh_t,
                                                      const float* __restrict__ bias,
                                                      const long long* __restrict__ length,
                                                      float* __restrict__ h_out,
                                                      u16* __restrict__ h_shared,
                                                      unsigned* __restrict__ cnt,
                                                      int T) {
  __shared__ u16   Atile[BATCH * HID_D];   // 32 KB  : h_prev (bf16), all hidden units
  __shared__ float Gates[BATCH * 128];     // 16 KB  : this WG's 128 gate columns
  __shared__ float Cst[BATCH * 32];        // 4 KB   : cell state for our 32 hidden units
  __shared__ float Hp[BATCH * 32];         // 4 KB   : previous h slice (for masking)

  const int tid = threadIdx.x;
  const int lane = tid & 31, wave = tid >> 5;
  const int hlf = lane >> 4, l16 = lane & 15;
  const int wg = blockIdx.x;               // 0..15
  const int hidbase = wg * 32;
  const unsigned nWG = gridDim.x;

  // init states + our slice of the global h exchange buffer
  for (int e = tid; e < BATCH * 32; e += 256) {
    Cst[e] = 0.0f; Hp[e] = 0.0f;
    int b = e >> 5, j = e & 31;
    h_shared[b * HID_D + hidbase + j] = 0;  // bf16 +0.0
  }
  __threadfence();
  __syncthreads();
  unsigned phase = 1;
  if (tid == 0) {
    atomicAdd(cnt, 1u);
    while (*(volatile unsigned*)cnt < nWG * phase) __builtin_amdgcn_s_sleep(1);
  }
  __syncthreads();

  // local gate column of this wave: cl in [0,128): gate = cl/32, hid = hidbase + cl%32
  const int cl = wave * 16 + l16;
  const int gc = (cl >> 5) * HID_D + hidbase + (cl & 31);   // global gate column
  const float bcol = bias[gc];
  const u16* Bcol = Whh_t + (size_t)gc * HID_D;
  const float* wcol = wiT + (size_t)gc * (T_STEPS * BATCH);

  // this thread's fixed batch row for the elementwise phase
  const int e0 = tid * 4;
  const int eb = e0 >> 5, ej = e0 & 31;
  const long long llen = length[eb];

#if __has_builtin(__builtin_amdgcn_tensor_load_to_lds)
  // --- Tensor Data Mover descriptor: h_shared (32 KB contiguous) -> Atile in LDS ---
  const unsigned lds_addr = (unsigned)(unsigned long long)(void*)&Atile[0];
  const unsigned long long ga = (unsigned long long)(const void*)h_shared;
  u32x4 g0;
  g0[0] = 1u;                                   // count=1, user descriptor
  g0[1] = lds_addr;                             // LDS byte address
  g0[2] = (unsigned)(ga & 0xFFFFFFFFull);       // global_addr[31:0]
  g0[3] = (unsigned)((ga >> 32) & 0x01FFFFFFull) | (2u << 30);  // [56:32] | type=2
  i32x8 g1;
  g1[0] = 0x00010000;                 // workgroup_mask=0, data_size=1 (2 bytes)
  g1[1] = (int)(16384u << 16);        // tensor_dim0[15:0]=16384 (elements)
  g1[2] = (int)(1u << 16);            // tensor_dim0[31:16]=0, tensor_dim1=1
  g1[3] = (int)(16384u << 16);        // tile_dim0 = 16384 elements (one row)
  g1[4] = 1;                          // tile_dim1=1, tile_dim2=0
  g1[5] = 16384;                      // tensor_dim0_stride[31:0]
  g1[6] = 0;
  g1[7] = 0;
  i32x4 z4 = {0, 0, 0, 0};
#if __clang_major__ >= 23
  i32x8 z8 = {0, 0, 0, 0, 0, 0, 0, 0};
#endif
#endif

  for (int t = 0; t < T; ++t) {
    const float* wp = wcol + t * BATCH;

#if __has_builtin(__builtin_amdgcn_tensor_load_to_lds)
    // stage h_prev (B x H bf16, 32 KB) into LDS via the Tensor Data Mover
    if (wave == 0) {
#if __clang_major__ >= 23
      __builtin_amdgcn_tensor_load_to_lds(g0, g1, z4, z4, z8, 0);
#else
      __builtin_amdgcn_tensor_load_to_lds(g0, g1, z4, z4, 0);
#endif
      __builtin_amdgcn_s_wait_tensorcnt(0);
    }
#else
    for (int e = tid * 8; e < BATCH * HID_D; e += 256 * 8)
      *(v8bf*)(Atile + e) = *(const v8bf*)(h_shared + e);
#endif
    // prefetch next timestep's wi slice while we compute this one
    if (t + 1 < T) __builtin_prefetch(wp + BATCH, 0, 1);
    __syncthreads();

    // gates = wi_t + b + h_prev @ Whh   (M=32 batch rows -> 2 accumulators)
    v8f acc0, acc1;
    {
      v4f w0 = *(const v4f*)(wp + hlf * 8);
      v4f w1 = *(const v4f*)(wp + hlf * 8 + 4);
      v4f w2 = *(const v4f*)(wp + 16 + hlf * 8);
      v4f w3 = *(const v4f*)(wp + 16 + hlf * 8 + 4);
#pragma unroll
      for (int i = 0; i < 4; ++i) {
        acc0[i]     = w0[i] + bcol;
        acc0[4 + i] = w1[i] + bcol;
        acc1[i]     = w2[i] + bcol;
        acc1[4 + i] = w3[i] + bcol;
      }
    }
#pragma unroll 4
    for (int kc = 0; kc < HID_D; kc += 32) {
      v8bf b0 = *(const v8bf*)(Bcol + kc + hlf * 8);
      v8bf b1 = *(const v8bf*)(Bcol + kc + 16 + hlf * 8);
      v16bf bf;
#pragma unroll
      for (int i = 0; i < 8; ++i) { bf[i] = b0[i]; bf[8 + i] = b1[i]; }

      v8bf x0 = *(const v8bf*)(Atile + l16 * HID_D + kc + hlf * 8);
      v8bf x1 = *(const v8bf*)(Atile + l16 * HID_D + kc + 16 + hlf * 8);
      v8bf y0 = *(const v8bf*)(Atile + (16 + l16) * HID_D + kc + hlf * 8);
      v8bf y1 = *(const v8bf*)(Atile + (16 + l16) * HID_D + kc + 16 + hlf * 8);
      v16bf af0, af1;
#pragma unroll
      for (int i = 0; i < 8; ++i) {
        af0[i] = x0[i]; af0[8 + i] = x1[i];
        af1[i] = y0[i]; af1[8 + i] = y1[i];
      }
      acc0 = __builtin_amdgcn_wmma_f32_16x16x32_bf16(false, af0, false, bf,
                                                     (short)0, acc0, false, false);
      acc1 = __builtin_amdgcn_wmma_f32_16x16x32_bf16(false, af1, false, bf,
                                                     (short)0, acc1, false, false);
    }

    // spill gates to LDS so the 4 gates of one hidden unit can be combined
#pragma unroll
    for (int r = 0; r < 8; ++r) {
      Gates[(hlf * 8 + r) * 128 + cl] = acc0[r];
      Gates[(16 + hlf * 8 + r) * 128 + cl] = acc1[r];
    }
    __syncthreads();

    // elementwise LSTM cell update + length mask (4 contiguous j per thread)
    {
      v4f fg = *(const v4f*)&Gates[eb * 128 + ej];
      v4f ig = *(const v4f*)&Gates[eb * 128 + 32 + ej];
      v4f og = *(const v4f*)&Gates[eb * 128 + 64 + ej];
      v4f gg = *(const v4f*)&Gates[eb * 128 + 96 + ej];
      v4f co = *(const v4f*)&Cst[eb * 32 + ej];
      v4f ho = *(const v4f*)&Hp[eb * 32 + ej];
      const bool m = (long long)t < llen;
      v4f c1, h1;
      u16x4 hb;
#pragma unroll
      for (int i = 0; i < 4; ++i) {
        float c = sigmf(fg[i]) * co[i] + sigmf(ig[i]) * tanhfast(gg[i]);
        float h = sigmf(og[i]) * tanhfast(c);
        c1[i] = m ? c : co[i];
        h1[i] = m ? h : ho[i];
        hb[i] = f2bf_u(h1[i]);
      }
      *(v4f*)&Cst[eb * 32 + ej] = c1;
      *(v4f*)&Hp[eb * 32 + ej] = h1;
      *(v4f*)(h_out + (size_t)t * BATCH * HID_D + (size_t)eb * HID_D + hidbase + ej) = h1;
      *(u16x4*)(h_shared + eb * HID_D + hidbase + ej) = hb;
    }
    __threadfence();
    __syncthreads();
    ++phase;
    if (tid == 0) {
      atomicAdd(cnt, 1u);
      while (*(volatile unsigned*)cnt < nWG * phase) __builtin_amdgcn_s_sleep(1);
    }
    __syncthreads();
  }
}

// ---------- host orchestration ----------
extern "C" void kernel_launch(void* const* d_in, const int* in_sizes, int n_in,
                              void* d_out, int out_size, void* d_ws, size_t ws_size,
                              hipStream_t stream) {
  (void)in_sizes; (void)n_in; (void)out_size; (void)ws_size;
  const float*     x      = (const float*)d_in[0];
  const long long* length = (const long long*)d_in[1];   // int64
  const float*     Wih0   = (const float*)d_in[2];
  const float*     Whh0   = (const float*)d_in[3];
  const float*     b0     = (const float*)d_in[4];
  const float*     Wih1   = (const float*)d_in[5];
  const float*     Whh1   = (const float*)d_in[6];
  const float*     b1     = (const float*)d_in[7];
  const float*     Wfc    = (const float*)d_in[8];
  const float*     bfc    = (const float*)d_in[9];
  float*           out    = (float*)d_out;

  char* ws = (char*)d_ws;
  size_t off = 0;
  auto alloc = [&](size_t bytes) -> char* {
    char* p = ws + off;
    off = (off + bytes + 255) & ~(size_t)255;
    return p;
  };
  u16* Wih0t = (u16*)alloc((size_t)GATE4 * FEAT_D * 2);
  u16* Whh0t = (u16*)alloc((size_t)GATE4 * HID_D * 2);
  u16* Wih1t = (u16*)alloc((size_t)GATE4 * HID_D * 2);
  u16* Whh1t = (u16*)alloc((size_t)GATE4 * HID_D * 2);
  u16* Wfct  = (u16*)alloc((size_t)ANCH * HID_D * 2);
  u16* hsh   = (u16*)alloc((size_t)BATCH * HID_D * 2);
  unsigned* cnts = (unsigned*)alloc(256);
  float* wibufT = (float*)alloc((size_t)T_STEPS * BATCH * GATE4 * 4);  // (4H, T*B)
  float* h0     = (float*)alloc((size_t)T_STEPS * BATCH * HID_D * 4);
  float* h1     = (float*)alloc((size_t)T_STEPS * BATCH * HID_D * 4);

  constexpr int M = T_STEPS * BATCH;  // 65536

  k_init_counters<<<1, 32, 0, stream>>>(cnts);
  k_transpose_to_bf16<<<512, 256, 0, stream>>>(Wih0, Wih0t, FEAT_D, GATE4);
  k_transpose_to_bf16<<<512, 256, 0, stream>>>(Whh0, Whh0t, HID_D, GATE4);
  k_transpose_to_bf16<<<512, 256, 0, stream>>>(Wih1, Wih1t, HID_D, GATE4);
  k_transpose_to_bf16<<<512, 256, 0, stream>>>(Whh1, Whh1t, HID_D, GATE4);
  k_transpose_to_bf16<<<128, 256, 0, stream>>>(Wfc, Wfct, HID_D, ANCH);

  // layer 0: wi0 = x @ Wih0, stored transposed (bias folded into the scan)
  k_gemm_bf16<M, GATE4, FEAT_D, true><<<(M / 16) * (GATE4 / 64) / 8, 256, 0, stream>>>(
      x, Wih0t, nullptr, wibufT);
  k_lstm_scan<<<16, 256, 0, stream>>>(wibufT, Whh0t, b0, length, h0, hsh, cnts + 0, T_STEPS);

  // layer 1: wi1 = h0 @ Wih1, stored transposed
  k_gemm_bf16<M, GATE4, HID_D, true><<<(M / 16) * (GATE4 / 64) / 8, 256, 0, stream>>>(
      h0, Wih1t, nullptr, wibufT);
  k_lstm_scan<<<16, 256, 0, stream>>>(wibufT, Whh1t, b1, length, h1, hsh, cnts + 1, T_STEPS);

  // final FC: out = h1 @ Wfc + bfc (row-major, matches d_out layout)
  k_gemm_bf16<M, ANCH, HID_D, false><<<(M / 16) * (ANCH / 64) / 8, 256, 0, stream>>>(
      h1, Wfct, bfc, out);
}
